// MultiHeadAttention_21964462752242
// MI455X (gfx1250) — matrix operand order repaired and checked by reading, not compiled, not run on hardware
//
#include <hip/hip_runtime.h>
#include <math.h>

// ---------------------------------------------------------------------------
// MultiHeadAttention forward for MI455X (gfx1250), bf16 WMMA pipeline.
//   1) convert x / Wq / Wk / Wv / Wo  f32 -> bf16              (cvt kernel)
//   2) Q = 0.125 * x Wq^T, K = x Wk^T (bf16, row-major)        (gemm mode 0)
//   3) V = x Wv^T stored TRANSPOSED as Vt[B,NH,HD,T]           (gemm mode 1)
//   4) flash attention (causal, online softmax), O bf16        (attn kernel)
//   5) out = O Wo^T + bo, f32                                  (gemm mode 2)
// All matmuls use v_wmma_f32_16x16x32_bf16 (wave32, f32 accum).
// ---------------------------------------------------------------------------

typedef __bf16 bf16;
typedef __bf16 v16bf __attribute__((ext_vector_type(16)));
typedef __bf16 v8bf  __attribute__((ext_vector_type(8)));
typedef __bf16 v4bf  __attribute__((ext_vector_type(4)));
typedef float  v8f   __attribute__((ext_vector_type(8)));

#define WMMA_BF16(a, b, c) \
  __builtin_amdgcn_wmma_f32_16x16x32_bf16(false, (a), false, (b), (short)0, (c), false, false)

static constexpr int Bsz = 2, T = 2048, D = 1024, NH = 16, HD = 64;
static constexpr int MROWS = Bsz * T;  // 4096

// ---------------------------------------------------------------- convert --
__global__ __launch_bounds__(256)
void cvt_f32_to_bf16_x4(const float* __restrict__ in, bf16* __restrict__ out, int n4) {
  int i = blockIdx.x * blockDim.x + threadIdx.x;
  if (i < n4) {
    float4 f = ((const float4*)in)[i];
    v4bf r;
    r.x = (bf16)f.x; r.y = (bf16)f.y; r.z = (bf16)f.z; r.w = (bf16)f.w;
    ((v4bf*)out)[i] = r;
  }
}

// ------------------------------------------------------------------- GEMM --
// y[M,N] = A[M,K] * W[N,K]^T.  K = N = D = 1024.
// Per block: 8 waves, each wave owns a 16x64 output slab. Grid (M/128, N/64).
// mode 0: bf16 store, scaled.   mode 1: store as Vt[B,NH,HD,T] (packed b128).
// mode 2: f32 store + bias (final projection into d_out).
__global__ __launch_bounds__(256)
void gemm_xwT(const bf16* __restrict__ A, const bf16* __restrict__ W,
              bf16* __restrict__ outB, float* __restrict__ outF,
              const float* __restrict__ bias, int mode, float scale) {
  const int lane = threadIdx.x & 31;
  const int wave = threadIdx.x >> 5;
  const int hi   = (lane >> 4) & 1;   // lane half (selects K sub-range)
  const int nl   = lane & 15;
  const int row0 = blockIdx.x * 128 + wave * 16;
  const int col0 = blockIdx.y * 64;

  v8f acc[4] = {};
  const bf16* aRow = A + (size_t)(row0 + nl) * D;  // this lane's A row (m = nl)
  const int aOff = hi ? 16 : 0;

  for (int k0 = 0; k0 < D; k0 += 32) {
    if (k0 + 128 < D) __builtin_prefetch(aRow + k0 + 128, 0, 1);
    union { v16bf v; v8bf h[2]; } a;
    a.h[0] = *(const v8bf*)(aRow + k0 + aOff);        // A 16x32: K 16h+0..7
    a.h[1] = *(const v8bf*)(aRow + k0 + 8 + aOff);    //          K 16h+8..15
#pragma unroll
    for (int c = 0; c < 4; ++c) {
      // B 32x16: lane n = nl holds W row (col0+c*16+nl), K contiguous.
      const bf16* wRow = W + (size_t)(col0 + c * 16 + nl) * D + k0 + (hi ? 16 : 0);
      v16bf b = *(const v16bf*)wRow;
      acc[c] = WMMA_BF16(a.v, b, acc[c]);
    }
  }

  if (mode == 0) {            // bf16 row-major, scaled (Q gets 1/sqrt(HD))
#pragma unroll
    for (int c = 0; c < 4; ++c) {
      int col = col0 + c * 16 + nl;
#pragma unroll
      for (int r = 0; r < 8; ++r) {
        int row = row0 + r + 8 * hi;
        outB[(size_t)row * D + col] = (bf16)(acc[c][r] * scale);
      }
    }
  } else if (mode == 1) {     // Vt[((b*NH+h)*HD+hd)*T + t], rows r are t-contig
    int b = row0 / T;
    int t0 = (row0 % T) + 8 * hi;
#pragma unroll
    for (int c = 0; c < 4; ++c) {
      int col = col0 + c * 16 + nl;   // d = h*HD + hd
      int h = col / HD, hd = col % HD;
      v8bf pk;
#pragma unroll
      for (int r = 0; r < 8; ++r) pk[r] = (bf16)acc[c][r];
      *(v8bf*)(outB + ((size_t)(b * NH + h) * HD + hd) * T + t0) = pk;
    }
  } else {                    // f32 + bias -> d_out
#pragma unroll
    for (int c = 0; c < 4; ++c) {
      int col = col0 + c * 16 + nl;
      float bb = bias[col];
#pragma unroll
      for (int r = 0; r < 8; ++r) {
        int row = row0 + r + 8 * hi;
        outF[(size_t)row * D + col] = acc[c][r] + bb;
      }
    }
  }
}

// -------------------------------------------------------- flash attention --
// Grid: (B*NH, T/128), 256 threads = 8 waves; each wave owns 16 query rows.
// Key blocks of 32 (= one bf16 WMMA K step). Scale pre-folded into Q.
__global__ __launch_bounds__(256)
void flash_attn_causal(const bf16* __restrict__ Q, const bf16* __restrict__ Kmat,
                       const bf16* __restrict__ Vt, bf16* __restrict__ O) {
  __shared__ bf16 pbuf[8][16][40];  // per-wave P relayout tile (stride 40 -> 16B rows)

  const int lane = threadIdx.x & 31;
  const int wave = threadIdx.x >> 5;
  const int hi = (lane >> 4) & 1, nl = lane & 15;
  const int bh = blockIdx.x;
  const int b = bh / NH, h = bh % NH;
  const int q0 = blockIdx.y * 128 + wave * 16;

  // Q A-fragments for HD=64 (two K-chunks of 32), kept live across key loop.
  const bf16* qRow = Q + ((size_t)(b * T + q0 + nl) * D + h * HD);
  const int aOff = hi ? 16 : 0;
  union Frag { v16bf v; v8bf h[2]; };
  Frag aQ[2];
#pragma unroll
  for (int kc = 0; kc < 2; ++kc) {
    aQ[kc].h[0] = *(const v8bf*)(qRow + kc * 32 + aOff);
    aQ[kc].h[1] = *(const v8bf*)(qRow + kc * 32 + 8 + aOff);
  }

  float mrow[8], lrow[8];
  v8f o[4] = {};
#pragma unroll
  for (int r = 0; r < 8; ++r) { mrow[r] = -1e30f; lrow[r] = 0.f; }

  const int kEnd = q0 + 16;  // causal: need keys kj <= q0+15
  for (int kb = 0; kb < kEnd; kb += 32) {
    // ---- S = (scaled Q) K^T : two 16-key subtiles, 4 wmma ----
    v8f s[2] = {};
#pragma unroll
    for (int sub = 0; sub < 2; ++sub) {
      const bf16* kRow =
          Kmat + ((size_t)(b * T + kb + sub * 16 + nl) * D + h * HD + (hi ? 16 : 0));
      v16bf b0 = *(const v16bf*)(kRow);       // hd 0..31 slice
      v16bf b1 = *(const v16bf*)(kRow + 32);  // hd 32..63 slice
      s[sub] = WMMA_BF16(aQ[0].v, b0, s[sub]);
      s[sub] = WMMA_BF16(aQ[1].v, b1, s[sub]);
    }

    // ---- causal mask (C layout: row = r+8*hi, col = nl) ----
#pragma unroll
    for (int sub = 0; sub < 2; ++sub)
#pragma unroll
      for (int r = 0; r < 8; ++r) {
        int qi = q0 + r + 8 * hi;
        int kj = kb + sub * 16 + nl;
        if (kj > qi) s[sub][r] = -3.0e38f;
      }

    // ---- online softmax: rowmax / rescale / rowsum (16-lane reductions) ----
    float mnew[8], alpha[8];
#pragma unroll
    for (int r = 0; r < 8; ++r) {
      float v = fmaxf(s[0][r], s[1][r]);
#pragma unroll
      for (int off = 8; off >= 1; off >>= 1) v = fmaxf(v, __shfl_xor(v, off, 16));
      mnew[r] = fmaxf(mrow[r], v);
      alpha[r] = __expf(mrow[r] - mnew[r]);
    }
#pragma unroll
    for (int r = 0; r < 8; ++r) {
      float p0 = __expf(s[0][r] - mnew[r]);
      float p1 = __expf(s[1][r] - mnew[r]);
      s[0][r] = p0; s[1][r] = p1;
      float rs = p0 + p1;
#pragma unroll
      for (int off = 8; off >= 1; off >>= 1) rs += __shfl_xor(rs, off, 16);
      lrow[r] = lrow[r] * alpha[r] + rs;
      mrow[r] = mnew[r];
    }
#pragma unroll
    for (int c = 0; c < 4; ++c)
#pragma unroll
      for (int r = 0; r < 8; ++r) o[c][r] *= alpha[r];

    // ---- P: C-layout -> LDS -> A-fragment relayout ----
#pragma unroll
    for (int sub = 0; sub < 2; ++sub)
#pragma unroll
      for (int r = 0; r < 8; ++r)
        pbuf[wave][r + 8 * hi][sub * 16 + nl] = (bf16)s[sub][r];
    asm volatile("s_wait_dscnt 0x0" ::: "memory");  // LDS RAW fence (per-wave tile)
    Frag aP;
    aP.h[0] = *(const v8bf*)(&pbuf[wave][nl][aOff]);
    aP.h[1] = *(const v8bf*)(&pbuf[wave][nl][8 + aOff]);

    // ---- O += P * V : Vt gives contiguous B-fragments, 4 wmma ----
#pragma unroll
    for (int c = 0; c < 4; ++c) {
      const bf16* vCol =
          Vt + ((size_t)(b * NH + h) * HD + c * 16 + nl) * T + kb + (hi ? 16 : 0);
      v16bf bv = *(const v16bf*)vCol;
      o[c] = WMMA_BF16(aP.v, bv, o[c]);
    }
  }

  // ---- normalize and store O (bf16, [B,T,D] with d = h*HD+hd) ----
  float inv[8];
#pragma unroll
  for (int r = 0; r < 8; ++r) inv[r] = 1.0f / lrow[r];
#pragma unroll
  for (int c = 0; c < 4; ++c) {
    int hd = c * 16 + nl;
#pragma unroll
    for (int r = 0; r < 8; ++r) {
      int t = q0 + r + 8 * hi;
      O[(size_t)(b * T + t) * D + h * HD + hd] = (bf16)(o[c][r] * inv[r]);
    }
  }
}

// ----------------------------------------------------------------- launch --
extern "C" void kernel_launch(void* const* d_in, const int* in_sizes, int n_in,
                              void* d_out, int out_size, void* d_ws, size_t ws_size,
                              hipStream_t stream) {
  const float* x  = (const float*)d_in[0];
  const float* Wq = (const float*)d_in[1];
  const float* Wk = (const float*)d_in[2];
  const float* Wv = (const float*)d_in[3];
  const float* Wo = (const float*)d_in[4];
  const float* bo = (const float*)d_in[5];
  float* out = (float*)d_out;

  // Workspace layout (bf16), 48 MB total.
  char* ws = (char*)d_ws;
  const size_t MB = 1024 * 1024;
  bf16* xb  = (bf16*)(ws + 0 * MB);   // 8 MB  [4096,1024]
  bf16* Wqb = (bf16*)(ws + 8 * MB);   // 2 MB
  bf16* Wkb = (bf16*)(ws + 10 * MB);  // 2 MB
  bf16* Wvb = (bf16*)(ws + 12 * MB);  // 2 MB
  bf16* Wob = (bf16*)(ws + 14 * MB);  // 2 MB
  bf16* Qb  = (bf16*)(ws + 16 * MB);  // 8 MB  [4096,1024] (pre-scaled by 0.125)
  bf16* Kb  = (bf16*)(ws + 24 * MB);  // 8 MB  [4096,1024]
  bf16* Vt  = (bf16*)(ws + 32 * MB);  // 8 MB  [B,NH,HD,T]
  bf16* Ob  = (bf16*)(ws + 40 * MB);  // 8 MB  [4096,1024]

  {
    int n4 = (Bsz * T * D) / 4;
    cvt_f32_to_bf16_x4<<<(n4 + 255) / 256, 256, 0, stream>>>(x, xb, n4);
  }
  {
    int n4 = (D * D) / 4;
    cvt_f32_to_bf16_x4<<<(n4 + 255) / 256, 256, 0, stream>>>(Wq, Wqb, n4);
    cvt_f32_to_bf16_x4<<<(n4 + 255) / 256, 256, 0, stream>>>(Wk, Wkb, n4);
    cvt_f32_to_bf16_x4<<<(n4 + 255) / 256, 256, 0, stream>>>(Wv, Wvb, n4);
    cvt_f32_to_bf16_x4<<<(n4 + 255) / 256, 256, 0, stream>>>(Wo, Wob, n4);
  }

  dim3 gg(MROWS / 128, D / 64);
  gemm_xwT<<<gg, 256, 0, stream>>>(xb, Wqb, Qb, nullptr, nullptr, 0, 0.125f);  // 1/sqrt(64)
  gemm_xwT<<<gg, 256, 0, stream>>>(xb, Wkb, Kb, nullptr, nullptr, 0, 1.0f);
  gemm_xwT<<<gg, 256, 0, stream>>>(xb, Wvb, Vt, nullptr, nullptr, 1, 1.0f);

  dim3 ga(Bsz * NH, T / 128);
  flash_attn_causal<<<ga, 256, 0, stream>>>(Qb, Kb, Vt, Ob);

  gemm_xwT<<<gg, 256, 0, stream>>>(Ob, Wob, nullptr, out, bo, 2, 1.0f);

  (void)in_sizes; (void)n_in; (void)out_size; (void)ws_size;
}
